// STGATEncoder_678604833047
// MI455X (gfx1250) — matrix operand order repaired and checked by reading, not compiled, not run on hardware
//
#include <hip/hip_runtime.h>
#include <hip/hip_bf16.h>

// STGAT encoder for MI455X (gfx1250, wave32, WMMA).
// - All GEMMs (LSTM gates, gat_fc, A_norm@Wh, both MLP heads) run on
//   v_wmma_f32_16x16x32_f16 (f16 in / f32 accumulate).
// - LSTM recurrent state lives in LDS; x_t is streamed in with
//   global_load_async_to_lds_b32 + s_wait_asynccnt (CDNA5 async path),
//   overlapped with the c/h elementwise update.
// - Activations use v_rcp_f32 / v_exp_f32 directly (no libm branches).
//
// Input leaf order assumes jax tree (sorted dict keys) flattening:
//  0:X  1:emb.b 2:emb.w  3:gat_a.b 4:gat_a.w  5:gat_fc.b 6:gat_fc.w
//  7:inter_mlp.fc1.b 8:.w 9:fc2.b 10:.w
//  11..26: interact_rnn {l_hf,l_hg,l_hi,l_ho,l_if,l_ig,l_ii,l_io} x {b,w}
//  27:motion_mlp.fc1.b 28:.w 29:fc2.b 30:.w
//  31..46: motion_rnn  {l_hf,l_hg,l_hi,l_ho,l_if,l_ig,l_ii,l_io} x {b,w}
//  47:rel_rec 48:rel_send  (unused: edge (i,j) computed analytically)

typedef __attribute__((ext_vector_type(16))) _Float16 v16h;
typedef __attribute__((ext_vector_type(8)))  _Float16 v8h;
typedef __attribute__((ext_vector_type(8)))  float    v8f;

#define BB   8
#define NN   48
#define TT   128
#define TP   127           // T-1
#define BN   384           // B*N

#define LOG2E 1.44269504088896340736f

static __device__ __forceinline__ v8f wmma_f16(v16h a, v16h b, v8f c) {
  return __builtin_amdgcn_wmma_f32_16x16x32_f16(false, a, false, b, (short)0, c,
                                                false, false);
}

// v_rcp_f32 / v_exp_f32 based activations (no IEEE-div, no libm branches)
static __device__ __forceinline__ float fast_sig(float x) {
  return __builtin_amdgcn_rcpf(1.0f + __builtin_amdgcn_exp2f(-LOG2E * x));
}
static __device__ __forceinline__ float fast_tanh(float x) {
  return 2.0f * fast_sig(2.0f * x) - 1.0f;
}
static __device__ __forceinline__ float fast_exp(float x) {
  return __builtin_amdgcn_exp2f(LOG2E * x);
}

// gfx1250 async global->LDS copy (ASYNCcnt tracked). LDS address VGPR takes
// the LDS byte offset == low 32 bits of the generic (flat) shared pointer.
static __device__ __forceinline__ void async_ld32(uint32_t ldsoff,
                                                  const void* gaddr) {
  asm volatile("global_load_async_to_lds_b32 %0, %1, off"
               :: "v"(ldsoff), "v"((uint64_t)(uintptr_t)gaddr) : "memory");
}
static __device__ __forceinline__ void wait_async0() {
  asm volatile("s_wait_asynccnt 0x0" ::: "memory");
}

// A-fragment (16xK f16, row-major rows with `stride` halves), K-chunk 0/1.
// Lanes 0-15: K = {0..7, 16..23}+32*chunk ; lanes 16-31: K = {8..15, 24..31}.
static __device__ __forceinline__ v16h load_a_frag(const _Float16* base,
                                                   int stride, int kchunk) {
  int lane = threadIdx.x & 31;
  int row  = lane & 15;
  int koff = ((lane >> 4) ? 8 : 0) + kchunk * 32;
  const _Float16* p = base + (size_t)row * stride + koff;
  v8h lo = *(const v8h*)(p);
  v8h hi = *(const v8h*)(p + 16);
  return __builtin_shufflevector(lo, hi, 0, 1, 2, 3, 4, 5, 6, 7,
                                 8, 9, 10, 11, 12, 13, 14, 15);
}

// B-fragment (32x16 f16) from column-major weights (col stride in halves).
static __device__ __forceinline__ v16h load_b_frag(const _Float16* base,
                                                   int col_stride, int kchunk) {
  int lane = threadIdx.x & 31;
  int col  = lane & 15;
  int koff = (lane >> 4) * 8 + kchunk * 32;
  const _Float16* p = base + (size_t)col * col_stride + koff;
  v8h lo = *(const v8h*)(p);
  v8h hi = *(const v8h*)(p + 16);
  return __builtin_shufflevector(lo, hi, 0, 1, 2, 3, 4, 5, 6, 7,
                                 8, 9, 10, 11, 12, 13, 14, 15);
}

// ---------------------------------------------------------------- pack params
struct GateP { const float *wx, *bx, *wh, *bh; };
struct LstmP { GateP g[4]; };  // gate order: i, f, g, o

static __device__ void pack_lstm(const LstmP& p, _Float16* w /*[128c][64k] cm*/,
                                 float* bias /*[128]*/, int tid, int nt) {
  for (int g = 0; g < 4; ++g) {
    const GateP& q = p.g[g];
    for (int idx = tid; idx < 32 * 64; idx += nt) {
      int n = idx >> 6, k = idx & 63;
      float v = 0.0f;
      if (k < 16)      v = q.wx[k * 32 + n];
      else if (k < 48) v = q.wh[(k - 16) * 32 + n];
      w[(size_t)(g * 32 + n) * 64 + k] = (_Float16)v;
    }
    for (int n = tid; n < 32; n += nt) bias[g * 32 + n] = q.bx[n] + q.bh[n];
  }
}

__global__ __launch_bounds__(256) void pack_kernel(
    LstmP mo, LstmP it, const float* __restrict__ gatw_in /*32x16*/,
    const float* __restrict__ mw1, const float* __restrict__ mw2,
    const float* __restrict__ mb2,
    const float* __restrict__ iw1, const float* __restrict__ iw2,
    _Float16* wm, float* bm, _Float16* wi, float* bi, _Float16* gatw_pk,
    _Float16* mw1p, _Float16* mw2p, float* mb2p,
    _Float16* iw1p, _Float16* iw2p) {
  int tid = threadIdx.x;
  pack_lstm(mo, wm, bm, tid, 256);
  pack_lstm(it, wi, bi, tid, 256);
  for (int idx = tid; idx < 16 * 32; idx += 256) {  // gat_fc: cm [16c][32k]
    int n = idx >> 5, k = idx & 31;
    gatw_pk[n * 32 + k] = (_Float16)gatw_in[k * 16 + n];
  }
  for (int idx = tid; idx < 64 * 32; idx += 256) {  // fc1: cm [64c][32k]
    int c = idx >> 5, k = idx & 31;
    mw1p[c * 32 + k] = (_Float16)mw1[k * 64 + c];
    iw1p[c * 32 + k] = (_Float16)iw1[k * 64 + c];
  }
  for (int idx = tid; idx < 32 * 64; idx += 256) {  // m.fc2: cm [32c][64k] pad
    int c = idx >> 6, k = idx & 63;
    mw2p[c * 64 + k] = (_Float16)(c < 24 ? mw2[k * 24 + c] : 0.0f);
  }
  for (int idx = tid; idx < 16 * 64; idx += 256) {  // i.fc2: cm [16c][64k]
    int c = idx >> 6, k = idx & 63;
    iw2p[c * 64 + k] = (_Float16)iw2[k * 16 + c];
  }
  for (int c = tid; c < 32; c += 256) mb2p[c] = c < 24 ? mb2[c] : 0.0f;
}

// ------------------------------------------------------------ motion embedding
__global__ __launch_bounds__(256) void emb_kernel(
    const float* __restrict__ X, const float* __restrict__ w /*4x16*/,
    const float* __restrict__ b, _Float16* __restrict__ em /*[BN,TP,16]*/) {
  int idx = blockIdx.x * 256 + threadIdx.x;  // (cell,t)
  if (idx >= BN * TP) return;
  int cell = idx / TP, t = idx % TP;
  const float* x0 = X + ((size_t)cell * TT + t) * 4;
  float d0 = x0[4] - x0[0], d1 = x0[5] - x0[1];
  float d2 = x0[6] - x0[2], d3 = x0[7] - x0[3];
  _Float16* o = em + (size_t)idx * 16;
#pragma unroll
  for (int f = 0; f < 16; ++f) {
    float v = b[f] + d0 * w[f] + d1 * w[16 + f] + d2 * w[32 + f] + d3 * w[48 + f];
    o[f] = (_Float16)v;
  }
}

// ------------------------------------------------------------------- LSTM core
__global__ __launch_bounds__(128) void lstm_kernel(
    const _Float16* __restrict__ xin /*[BN,T,16]*/,
    _Float16* __restrict__ hout /*[BN,T,32]*/,
    const _Float16* __restrict__ wpk /*[128c][64k] cm*/,
    const float* __restrict__ bias /*[128]*/, int T) {
  __shared__ _Float16 a_buf[16 * 64];     // rows: K = [x(16) | h(32) | pad(16)]
  __shared__ float    gbuf[4][16 * 32];   // activated gates
  __shared__ float    cbuf[16 * 32];      // cell state

  int tid = threadIdx.x, lane = tid & 31, wave = tid >> 5;
  int r0 = blockIdx.x * 16;

  for (int idx = tid; idx < 16 * 64; idx += 128)
    if ((idx & 63) >= 16) a_buf[idx] = (_Float16)0.0f;  // h=0, pad=0
  for (int idx = tid; idx < 512; idx += 128) cbuf[idx] = 0.0f;

  const _Float16* wbase = wpk + (size_t)(wave * 32) * 64;
  v16h b00 = load_b_frag(wbase, 64, 0);
  v16h b01 = load_b_frag(wbase, 64, 1);
  v16h b10 = load_b_frag(wbase + (size_t)16 * 64, 64, 0);
  v16h b11 = load_b_frag(wbase + (size_t)16 * 64, 64, 1);
  float bias0 = bias[wave * 32 + (lane & 15)];
  float bias1 = bias[wave * 32 + 16 + (lane & 15)];
  bool is_g = (wave == 2);

  // per-thread async stage slot: 2 halves (4B); 128 threads cover 16x16
  int sidx = tid << 1;
  int srow = sidx >> 4, sk = sidx & 15;
  const _Float16* gsrc = xin + ((size_t)(r0 + srow) * T) * 16 + sk;  // t = 0
  uint32_t ldsdst = (uint32_t)(uintptr_t)&a_buf[srow * 64 + sk];

  async_ld32(ldsdst, gsrc);   // prologue: stage x(0)
  wait_async0();
  __syncthreads();

  for (int t = 0; t < T; ++t) {
    v16h a0 = load_a_frag(a_buf, 64, 0);
    v16h a1 = load_a_frag(a_buf, 64, 1);
    v8f acc0, acc1;
#pragma unroll
    for (int i = 0; i < 8; ++i) { acc0[i] = bias0; acc1[i] = bias1; }
    acc0 = wmma_f16(a0, b00, acc0);
    acc0 = wmma_f16(a1, b01, acc0);
    acc1 = wmma_f16(a0, b10, acc1);
    acc1 = wmma_f16(a1, b11, acc1);

    int n = lane & 15, mb = (lane >> 4) * 8;
    if (is_g) {  // wave-uniform branch: no per-element exec juggling
#pragma unroll
      for (int i = 0; i < 8; ++i) {
        gbuf[2][(mb + i) * 32 + n]      = fast_tanh(acc0[i]);
        gbuf[2][(mb + i) * 32 + 16 + n] = fast_tanh(acc1[i]);
      }
    } else {
#pragma unroll
      for (int i = 0; i < 8; ++i) {
        gbuf[wave][(mb + i) * 32 + n]      = fast_sig(acc0[i]);
        gbuf[wave][(mb + i) * 32 + 16 + n] = fast_sig(acc1[i]);
      }
    }
    __syncthreads();

    if (t + 1 < T) async_ld32(ldsdst, gsrc + (size_t)(t + 1) * 16);

    for (int idx = tid; idx < 512; idx += 128) {
      int row = idx >> 5, u = idx & 31;
      float iv = gbuf[0][idx], fv = gbuf[1][idx];
      float gv = gbuf[2][idx], ov = gbuf[3][idx];
      float c = fv * cbuf[idx] + iv * gv;
      cbuf[idx] = c;
      float h = ov * fast_tanh(c);
      a_buf[row * 64 + 16 + u] = (_Float16)h;
      hout[((size_t)(r0 + row) * T + t) * 32 + u] = (_Float16)h;
    }
    wait_async0();
    __syncthreads();
  }
}

// ------------------------------------------------- Wh = hm @ gat_fc (32 -> 16)
__global__ __launch_bounds__(32) void wh_kernel(
    const _Float16* __restrict__ hm /*[BN,TP,32]*/,
    const _Float16* __restrict__ gatw /*[16c][32k] cm*/,
    const float* __restrict__ gatb /*[16]*/,
    _Float16* __restrict__ wh_cm /*[B,TP,16,64]*/) {
  int blk = blockIdx.x;
  int b = blk / (3 * TP), rem = blk % (3 * TP);
  int ntile = rem / TP, t = rem % TP;
  int lane = threadIdx.x;
  int cell0 = b * NN + ntile * 16;

  const _Float16* abase = hm + ((size_t)cell0 * TP + t) * 32;
  v16h a = load_a_frag(abase, TP * 32, 0);
  v16h bf = load_b_frag(gatw, 32, 0);
  int n = lane & 15;
  float bv = gatb[n];
  v8f acc;
#pragma unroll
  for (int i = 0; i < 8; ++i) acc[i] = bv;
  acc = wmma_f16(a, bf, acc);

  _Float16* out = wh_cm + ((size_t)(b * TP + t) * 16) * 64;
  int mb = (lane >> 4) * 8;
#pragma unroll
  for (int i = 0; i < 8; ++i)
    out[n * 64 + (ntile * 16 + mb + i)] = (_Float16)acc[i];
  if (ntile == 0) {
    for (int idx = lane; idx < 16 * 16; idx += 32)
      out[(idx >> 4) * 64 + 48 + (idx & 15)] = (_Float16)0.0f;
  }
}

// ---------------------------------------- edge scores -> A[b,t,i,j] (diag = 0)
__global__ __launch_bounds__(256) void score_kernel(
    const _Float16* __restrict__ wh_cm, const float* __restrict__ aw /*[32]*/,
    const float* __restrict__ ab /*[1]*/, float* __restrict__ A) {
  int blk = blockIdx.x;
  int bt = blk / 9, part = blk % 9;
  int ij = part * 256 + threadIdx.x;
  int i = ij / NN, j = ij % NN;
  float* Abt = A + (size_t)bt * NN * NN;
  if (i == j) { Abt[ij] = 0.0f; return; }
  const _Float16* whb = wh_cm + (size_t)bt * 16 * 64;
  float s = ab[0];
#pragma unroll
  for (int f = 0; f < 16; ++f) {
    s += aw[f]      * (float)whb[f * 64 + i];
    s += aw[16 + f] * (float)whb[f * 64 + j];
  }
  float lr = s > 0.0f ? s : 0.2f * s;
  Abt[ij] = fast_exp(lr);
}

// ------------------------------------- A_sym, A_norm (+f16 padded for WMMA)
__global__ __launch_bounds__(256) void anorm_kernel(
    const float* __restrict__ A, float* __restrict__ A_sym,
    float* __restrict__ A_norm, _Float16* __restrict__ an16 /*[B,TP,48,64]*/) {
  __shared__ float tile[NN * NN];
  __shared__ float rs[NN];
  size_t base = (size_t)blockIdx.x * NN * NN;
  for (int idx = threadIdx.x; idx < NN * NN; idx += 256) tile[idx] = A[base + idx];
  __syncthreads();
  if (threadIdx.x < NN) {
    float s = 0.0f;
    for (int j = 0; j < NN; ++j) s += tile[threadIdx.x * NN + j];
    rs[threadIdx.x] = __builtin_amdgcn_rcpf(s);
  }
  __syncthreads();
  _Float16* anb = an16 + (size_t)blockIdx.x * NN * 64;
  for (int idx = threadIdx.x; idx < NN * NN; idx += 256) {
    int i = idx / NN, j = idx % NN;
    float a = tile[idx];
    A_sym[base + idx] = 0.5f * (a + tile[j * NN + i]);
    float an = a * rs[i];
    A_norm[base + idx] = an;
    anb[i * 64 + j] = (_Float16)an;
  }
  for (int idx = threadIdx.x; idx < NN * 16; idx += 256)
    anb[(idx / 16) * 64 + 48 + (idx % 16)] = (_Float16)0.0f;
}

// ------------------------------- em_p = relu(A_norm @ Wh_t), WMMA K=48 pad 64
__global__ __launch_bounds__(32) void hp_kernel(
    const _Float16* __restrict__ an16, const _Float16* __restrict__ wh_cm,
    _Float16* __restrict__ em_p /*[BN,TP,16]*/) {
  int blk = blockIdx.x;
  int bt = blk / 3, mtile = blk % 3;
  int b = bt / TP, t = bt % TP;
  const _Float16* abase = an16 + (size_t)bt * NN * 64 + (size_t)mtile * 16 * 64;
  const _Float16* bbase = wh_cm + (size_t)bt * 16 * 64;
  v16h a0 = load_a_frag(abase, 64, 0);
  v16h a1 = load_a_frag(abase, 64, 1);
  v16h b0 = load_b_frag(bbase, 64, 0);
  v16h b1 = load_b_frag(bbase, 64, 1);
  v8f acc = {};
  acc = wmma_f16(a0, b0, acc);
  acc = wmma_f16(a1, b1, acc);
  int lane = threadIdx.x, n = lane & 15, mb = (lane >> 4) * 8;
#pragma unroll
  for (int i = 0; i < 8; ++i) {
    int node = mtile * 16 + mb + i;
    float v = acc[i];
    v = v > 0.0f ? v : 0.0f;
    em_p[((size_t)(b * NN + node) * TP + t) * 16 + n] = (_Float16)v;
  }
}

// --------------------------------------- MLP heads (WMMA) + concat to output
__global__ __launch_bounds__(128) void head_kernel(
    const _Float16* __restrict__ hm, const _Float16* __restrict__ h2,
    const _Float16* __restrict__ mw1p, const float* __restrict__ mb1,
    const _Float16* __restrict__ mw2p, const float* __restrict__ mb2p,
    const _Float16* __restrict__ iw1p, const float* __restrict__ ib1,
    const _Float16* __restrict__ iw2p, const float* __restrict__ ib2,
    float* __restrict__ out) {
  __shared__ _Float16 hidm[16 * 64];
  __shared__ _Float16 hidi[16 * 64];
  int tid = threadIdx.x, lane = tid & 31, wave = tid >> 5;
  int r0 = blockIdx.x * 16;
  int n = lane & 15, mb = (lane >> 4) * 8;

  const _Float16* am_src = hm + ((size_t)r0 * TP + (TP - 1)) * 32;
  const _Float16* ai_src = h2 + ((size_t)r0 * TP + (TP - 1)) * 32;
  v16h am = load_a_frag(am_src, TP * 32, 0);
  v16h ai = load_a_frag(ai_src, TP * 32, 0);

  {  // motion fc1, N-tile `wave`
    v16h bf = load_b_frag(mw1p + (size_t)(wave * 16) * 32, 32, 0);
    v8f acc; float bv = mb1[wave * 16 + n];
#pragma unroll
    for (int i = 0; i < 8; ++i) acc[i] = bv;
    acc = wmma_f16(am, bf, acc);
#pragma unroll
    for (int i = 0; i < 8; ++i) {
      float v = acc[i]; v = v > 0.0f ? v : 0.0f;
      hidm[(mb + i) * 64 + wave * 16 + n] = (_Float16)v;
    }
  }
  {  // interact fc1, N-tile `wave`
    v16h bf = load_b_frag(iw1p + (size_t)(wave * 16) * 32, 32, 0);
    v8f acc; float bv = ib1[wave * 16 + n];
#pragma unroll
    for (int i = 0; i < 8; ++i) acc[i] = bv;
    acc = wmma_f16(ai, bf, acc);
#pragma unroll
    for (int i = 0; i < 8; ++i) {
      float v = acc[i]; v = v > 0.0f ? v : 0.0f;
      hidi[(mb + i) * 64 + wave * 16 + n] = (_Float16)v;
    }
  }
  __syncthreads();

  if (wave < 2) {  // motion fc2, out cols [wave*16, wave*16+16) of 24
    v16h a0 = load_a_frag(hidm, 64, 0);
    v16h a1 = load_a_frag(hidm, 64, 1);
    v16h b0 = load_b_frag(mw2p + (size_t)(wave * 16) * 64, 64, 0);
    v16h b1 = load_b_frag(mw2p + (size_t)(wave * 16) * 64, 64, 1);
    v8f acc; float bv = mb2p[wave * 16 + n];
#pragma unroll
    for (int i = 0; i < 8; ++i) acc[i] = bv;
    acc = wmma_f16(a0, b0, acc);
    acc = wmma_f16(a1, b1, acc);
    int col = wave * 16 + n;
#pragma unroll
    for (int i = 0; i < 8; ++i) {
      float v = acc[i]; v = v > 0.0f ? v : 0.0f;
      if (col < 24) out[(size_t)(r0 + mb + i) * 40 + col] = v;
    }
  } else if (wave == 2) {  // interact fc2, 16 out cols
    v16h a0 = load_a_frag(hidi, 64, 0);
    v16h a1 = load_a_frag(hidi, 64, 1);
    v16h b0 = load_b_frag(iw2p, 64, 0);
    v16h b1 = load_b_frag(iw2p, 64, 1);
    v8f acc; float bv = ib2[n];
#pragma unroll
    for (int i = 0; i < 8; ++i) acc[i] = bv;
    acc = wmma_f16(a0, b0, acc);
    acc = wmma_f16(a1, b1, acc);
#pragma unroll
    for (int i = 0; i < 8; ++i) {
      float v = acc[i]; v = v > 0.0f ? v : 0.0f;
      out[(size_t)(r0 + mb + i) * 40 + 24 + n] = v;
    }
  }
}

// ---------------------------------------------------------------------- launch
static inline size_t align256(size_t x) { return (x + 255) & ~(size_t)255; }

extern "C" void kernel_launch(void* const* d_in, const int* in_sizes, int n_in,
                              void* d_out, int out_size, void* d_ws,
                              size_t ws_size, hipStream_t stream) {
  const float* X    = (const float*)d_in[0];
  const float* embB = (const float*)d_in[1];
  const float* embW = (const float*)d_in[2];
  const float* gaB  = (const float*)d_in[3];
  const float* gaW  = (const float*)d_in[4];
  const float* gfB  = (const float*)d_in[5];
  const float* gfW  = (const float*)d_in[6];
  const float* iM1b = (const float*)d_in[7];
  const float* iM1w = (const float*)d_in[8];
  const float* iM2b = (const float*)d_in[9];
  const float* iM2w = (const float*)d_in[10];
  const float* I[16]; for (int k = 0; k < 16; ++k) I[k] = (const float*)d_in[11 + k];
  const float* mM1b = (const float*)d_in[27];
  const float* mM1w = (const float*)d_in[28];
  const float* mM2b = (const float*)d_in[29];
  const float* mM2w = (const float*)d_in[30];
  const float* M[16]; for (int k = 0; k < 16; ++k) M[k] = (const float*)d_in[31 + k];

  LstmP mo, it;
  mo.g[0] = { M[13], M[12], M[5],  M[4]  };  // i: ii, hi
  mo.g[1] = { M[9],  M[8],  M[1],  M[0]  };  // f: if, hf
  mo.g[2] = { M[11], M[10], M[3],  M[2]  };  // g: ig, hg
  mo.g[3] = { M[15], M[14], M[7],  M[6]  };  // o: io, ho
  it.g[0] = { I[13], I[12], I[5],  I[4]  };
  it.g[1] = { I[9],  I[8],  I[1],  I[0]  };
  it.g[2] = { I[11], I[10], I[3],  I[2]  };
  it.g[3] = { I[15], I[14], I[7],  I[6]  };

  char* ws = (char*)d_ws;
  size_t off = 0;
  auto carve = [&](size_t bytes) { char* p = ws + off; off += align256(bytes); return p; };
  _Float16* em    = (_Float16*)carve((size_t)BN * TP * 16 * 2);
  _Float16* hm    = (_Float16*)carve((size_t)BN * TP * 32 * 2);
  _Float16* h2    = (_Float16*)carve((size_t)BN * TP * 32 * 2);
  _Float16* wh_cm = (_Float16*)carve((size_t)BB * TP * 16 * 64 * 2);
  _Float16* an16  = (_Float16*)carve((size_t)BB * TP * NN * 64 * 2);
  _Float16* em_p  = (_Float16*)carve((size_t)BN * TP * 16 * 2);
  _Float16* wm    = (_Float16*)carve(128 * 64 * 2);
  float*    bm    = (float*)carve(128 * 4);
  _Float16* wi    = (_Float16*)carve(128 * 64 * 2);
  float*    bi    = (float*)carve(128 * 4);
  _Float16* gpk   = (_Float16*)carve(16 * 32 * 2);
  _Float16* mw1p  = (_Float16*)carve(64 * 32 * 2);
  _Float16* mw2p  = (_Float16*)carve(32 * 64 * 2);
  float*    mb2p  = (float*)carve(32 * 4);
  _Float16* iw1p  = (_Float16*)carve(64 * 32 * 2);
  _Float16* iw2p  = (_Float16*)carve(16 * 64 * 2);
  (void)ws_size; (void)n_in; (void)in_sizes; (void)out_size;

  float* out_h  = (float*)d_out;                       // [384,40]
  float* A      = out_h + (size_t)BN * 40;             // [8,127,48,48]
  float* A_sym  = A + (size_t)BB * TP * NN * NN;
  float* A_norm = A_sym + (size_t)BB * TP * NN * NN;

  pack_kernel<<<1, 256, 0, stream>>>(mo, it, gfW, mM1w, mM2w, mM2b, iM1w, iM2w,
                                     wm, bm, wi, bi, gpk,
                                     mw1p, mw2p, mb2p, iw1p, iw2p);
  emb_kernel<<<(BN * TP + 255) / 256, 256, 0, stream>>>(X, embW, embB, em);
  lstm_kernel<<<BN / 16, 128, 0, stream>>>(em, hm, wm, bm, TP);
  wh_kernel<<<BB * 3 * TP, 32, 0, stream>>>(hm, gpk, gfB, wh_cm);
  score_kernel<<<BB * TP * 9, 256, 0, stream>>>(wh_cm, gaW, gaB, A);
  anorm_kernel<<<BB * TP, 256, 0, stream>>>(A, A_sym, A_norm, an16);
  hp_kernel<<<BB * TP * 3, 32, 0, stream>>>(an16, wh_cm, em_p);
  lstm_kernel<<<BN / 16, 128, 0, stream>>>(em_p, h2, wi, bi, TP);
  head_kernel<<<BN / 16, 128, 0, stream>>>(hm, h2, mw1p, mM1b, mw2p, mb2p,
                                           iw1p, iM1b, iw2p, iM2b, out_h);
}
